// SegmentFFN_65764539236544
// MI455X (gfx1250) — compile-verified, matrix operand order checked by reading, not run on hardware
//
#include <hip/hip_runtime.h>
#include <hip/hip_bf16.h>

// ---------------------------------------------------------------------------
// Segmented SwiGLU FFN for MI455X (gfx1250, wave32, WMMA).
//   B=8, N=2048, D=1024, H=2048, 4 segments of 512 tokens.
//   kernel 1: h = silu(x@w1) * (x@w3)   (bf16 intermediate in d_ws)
//   kernel 2: out = h @ w2              (fp32 output)
// Both are 128x128x32 LDS-tiled GEMMs on v_wmma_f32_16x16x32_bf16.
// Kernel 2 stages its (already-bf16) A tile with GLOBAL_LOAD_ASYNC_TO_LDS.
// ---------------------------------------------------------------------------

typedef __attribute__((ext_vector_type(16))) __bf16 v16bf;
typedef __attribute__((ext_vector_type(8)))  float  v8f;

#define Dm   1024
#define Hm   2048
#define Ntok 2048
#define SEGL 512

#define BM 128
#define BN 128
#define BK 32
#define KP 40   // LDS halfword row stride: 80B -> 16B aligned frags, 20-bank stride

struct U32x8 { uint4 a, b; };

// round-to-nearest-even fp32 -> bf16 (as raw ushort)
static __device__ __forceinline__ unsigned short f2bf(float f) {
  unsigned u = __builtin_bit_cast(unsigned, f);
  unsigned r = u + 0x7FFFu + ((u >> 16) & 1u);
  return (unsigned short)(r >> 16);
}

// pack two fp32 into one dword of two bf16 (lo = a, hi = b)
static __device__ __forceinline__ unsigned pack_bf2(float a, float b) {
  return (unsigned)f2bf(a) | ((unsigned)f2bf(b) << 16);
}

// Build a 16xbf16 WMMA fragment from two 16B LDS chunks (ISA 16-bit A/B layout:
// this lane's 16 elements are k = 8*lh + {0..7} and k = 16 + 8*lh + {0..7}).
static __device__ __forceinline__ v16bf frag16(const unsigned short* p0,
                                               const unsigned short* p1) {
  U32x8 t;
  t.a = *reinterpret_cast<const uint4*>(p0);
  t.b = *reinterpret_cast<const uint4*>(p1);
  return __builtin_bit_cast(v16bf, t);
}

// token index for (segment, flat row within segment's 4096-row block)
static __device__ __forceinline__ long tok_of(int seg, int gr) {
  int b = gr >> 9;          // 512 tokens per (b, segment)
  int l = gr & 511;
  return (long)b * Ntok + (long)seg * SEGL + l;
}

// silu via v_exp_f32 + v_rcp_f32 (avoids IEEE div_scale/div_fixup expansion)
static __device__ __forceinline__ float silu_fast(float a) {
  return a * __builtin_amdgcn_rcpf(1.0f + __expf(-a));
}

// Async DMA: 16 bytes per lane, global -> LDS, tracked by ASYNCcnt.
// lds_off = wave-relative LDS byte offset (low 32 bits of generic pointer,
// per CDNA5 ISA 10.2 aperture rule).
static __device__ __forceinline__ void async_copy_b128(void* lds_ptr,
                                                       const void* gptr) {
  unsigned lds_off = (unsigned)(unsigned long long)lds_ptr;
  asm volatile("global_load_async_to_lds_b128 %0, %1, off"
               :: "v"(lds_off), "v"(gptr) : "memory");
}
static __device__ __forceinline__ void async_wait0() {
  asm volatile("s_wait_asynccnt 0x0" ::: "memory");
}

// ---------------------------------------------------------------------------
// Kernel 1: gate/up GEMMs + SwiGLU.  grid = (H/BN=16, 4096/BM=32, 4 segs)
// ---------------------------------------------------------------------------
__global__ __launch_bounds__(256) void seg_ffn_gate_kernel(
    const float* __restrict__ x, const float* __restrict__ w1,
    const float* __restrict__ w3, unsigned short* __restrict__ h) {
  __shared__ unsigned short As[BM * KP];
  __shared__ unsigned short B1s[BN * KP];
  __shared__ unsigned short B3s[BN * KP];

  const int tid  = threadIdx.x;
  const int lane = tid & 31;
  const int wm   = ((tid >> 5) & 1) * 64;   // wave M offset (2 groups of 64)
  const int wn   = ((tid >> 5) >> 1) * 32;  // wave N offset (4 groups of 32)
  const int seg   = blockIdx.z;
  const int ntile = blockIdx.x * BN;
  const int mtile = blockIdx.y * BM;

  const float* w1b = w1 + (long)seg * Dm * Hm;
  const float* w3b = w3 + (long)seg * Dm * Hm;

  v8f acc1[4][2], acc3[4][2];
  const v8f vz = {0.f, 0.f, 0.f, 0.f, 0.f, 0.f, 0.f, 0.f};
#pragma unroll
  for (int i = 0; i < 4; ++i)
#pragma unroll
    for (int j = 0; j < 2; ++j) { acc1[i][j] = vz; acc3[i][j] = vz; }

  // loader index precompute
  const int a_k = (tid & 7) * 4;   // A: k offset, 4 floats
  const int a_r = tid >> 3;        // A: row 0..31 (x4 passes)
  const int b_n = (tid & 31) * 4;  // B: n offset, 4 floats
  const int b_kp = tid >> 5;       // B: k-pair 0..7 (x2 passes)

  const int lh = lane >> 4;        // K-half select per ISA fragment layout
  const int lr = lane & 15;

  // hoist token->address math out of the K loop
  const float* a_base[4];
#pragma unroll
  for (int p = 0; p < 4; ++p)
    a_base[p] = x + tok_of(seg, mtile + a_r + 32 * p) * Dm + a_k;
  const float* b1_base = w1b + ntile + b_n;
  const float* b3_base = w3b + ntile + b_n;

  for (int kt = 0; kt < Dm; kt += BK) {
    // ---- stage A (x) fp32 -> bf16 into LDS -------------------------------
#pragma unroll
    for (int p = 0; p < 4; ++p) {
      int row = a_r + 32 * p;
      const float* gp = a_base[p] + kt;
      float4 v = *reinterpret_cast<const float4*>(gp);
      __builtin_prefetch(gp + BK, 0, 1);  // next K tile -> global_prefetch_b8
      *reinterpret_cast<unsigned*>(&As[row * KP + a_k])     = pack_bf2(v.x, v.y);
      *reinterpret_cast<unsigned*>(&As[row * KP + a_k + 2]) = pack_bf2(v.z, v.w);
    }
    // ---- stage B (w1, w3): two adjacent K rows, pack pairs, b32 stores ---
#pragma unroll
    for (int p = 0; p < 2; ++p) {
      int k0 = 2 * (b_kp + 8 * p);
      const float* g1 = b1_base + (long)(kt + k0) * Hm;
      const float* g3 = b3_base + (long)(kt + k0) * Hm;
      float4 u1 = *reinterpret_cast<const float4*>(g1);
      float4 v1 = *reinterpret_cast<const float4*>(g1 + Hm);
      float4 u3 = *reinterpret_cast<const float4*>(g3);
      float4 v3 = *reinterpret_cast<const float4*>(g3 + Hm);
      __builtin_prefetch(g1 + (long)BK * Hm, 0, 1);
      __builtin_prefetch(g3 + (long)BK * Hm, 0, 1);
      *reinterpret_cast<unsigned*>(&B1s[(b_n + 0) * KP + k0]) = pack_bf2(u1.x, v1.x);
      *reinterpret_cast<unsigned*>(&B1s[(b_n + 1) * KP + k0]) = pack_bf2(u1.y, v1.y);
      *reinterpret_cast<unsigned*>(&B1s[(b_n + 2) * KP + k0]) = pack_bf2(u1.z, v1.z);
      *reinterpret_cast<unsigned*>(&B1s[(b_n + 3) * KP + k0]) = pack_bf2(u1.w, v1.w);
      *reinterpret_cast<unsigned*>(&B3s[(b_n + 0) * KP + k0]) = pack_bf2(u3.x, v3.x);
      *reinterpret_cast<unsigned*>(&B3s[(b_n + 1) * KP + k0]) = pack_bf2(u3.y, v3.y);
      *reinterpret_cast<unsigned*>(&B3s[(b_n + 2) * KP + k0]) = pack_bf2(u3.z, v3.z);
      *reinterpret_cast<unsigned*>(&B3s[(b_n + 3) * KP + k0]) = pack_bf2(u3.w, v3.w);
    }
    __syncthreads();

    // ---- WMMA: 4x2 tiles x 2 GEMMs = 16 v_wmma per wave per K step -------
    v16bf af[4], bf1[2], bf3[2];
#pragma unroll
    for (int tm = 0; tm < 4; ++tm) {
      int row = wm + 16 * tm + lr;
      af[tm] = frag16(&As[row * KP + 8 * lh], &As[row * KP + 16 + 8 * lh]);
    }
#pragma unroll
    for (int tn = 0; tn < 2; ++tn) {
      int col = wn + 16 * tn + lr;
      bf1[tn] = frag16(&B1s[col * KP + 8 * lh], &B1s[col * KP + 16 + 8 * lh]);
      bf3[tn] = frag16(&B3s[col * KP + 8 * lh], &B3s[col * KP + 16 + 8 * lh]);
    }
#pragma unroll
    for (int tm = 0; tm < 4; ++tm)
#pragma unroll
      for (int tn = 0; tn < 2; ++tn) {
        acc1[tm][tn] = __builtin_amdgcn_wmma_f32_16x16x32_bf16(
            false, af[tm], false, bf1[tn], (short)0, acc1[tm][tn], false, false);
        acc3[tm][tn] = __builtin_amdgcn_wmma_f32_16x16x32_bf16(
            false, af[tm], false, bf3[tn], (short)0, acc3[tm][tn], false, false);
      }
    __syncthreads();
  }

  // ---- epilogue: h = silu(acc1) * acc3, store bf16 -----------------------
#pragma unroll
  for (int tm = 0; tm < 4; ++tm)
#pragma unroll
    for (int tn = 0; tn < 2; ++tn) {
      int colg = ntile + wn + 16 * tn + lr;
#pragma unroll
      for (int r = 0; r < 8; ++r) {
        int gr = mtile + wm + 16 * tm + 8 * lh + r;  // C layout: M = 8*lh + r
        long tok = tok_of(seg, gr);
        float a = acc1[tm][tn][r];
        float g = acc3[tm][tn][r];
        h[tok * Hm + colg] = f2bf(silu_fast(a) * g);
      }
    }
}

// ---------------------------------------------------------------------------
// Kernel 2: down GEMM.  grid = (D/BN=8, 4096/BM=32, 4 segs)
// A tile is staged via async global->LDS DMA (ASYNCcnt), overlapped with the
// synchronous fp32->bf16 B-tile staging.
// ---------------------------------------------------------------------------
__global__ __launch_bounds__(256) void seg_ffn_down_kernel(
    const unsigned short* __restrict__ h, const float* __restrict__ w2,
    float* __restrict__ out) {
  __shared__ unsigned short As[BM * KP];
  __shared__ unsigned short Bs[BN * KP];

  const int tid  = threadIdx.x;
  const int lane = tid & 31;
  const int wm   = ((tid >> 5) & 1) * 64;
  const int wn   = ((tid >> 5) >> 1) * 32;
  const int seg   = blockIdx.z;
  const int ntile = blockIdx.x * BN;
  const int mtile = blockIdx.y * BM;

  const float* w2b = w2 + (long)seg * Hm * Dm;

  v8f acc[4][2];
  const v8f vz = {0.f, 0.f, 0.f, 0.f, 0.f, 0.f, 0.f, 0.f};
#pragma unroll
  for (int i = 0; i < 4; ++i)
#pragma unroll
    for (int j = 0; j < 2; ++j) acc[i][j] = vz;

  const int a_k = (tid & 3) * 8;   // A: 8 bf16 per 16B async copy
  const int a_r = tid >> 2;        // A: row 0..63 (x2 passes)
  const int b_n = (tid & 31) * 4;  // B: n offset
  const int b_kp = tid >> 5;       // B: k-pair 0..7 (x2 passes)

  const int lh = lane >> 4;
  const int lr = lane & 15;

  const unsigned short* a_base[2];
#pragma unroll
  for (int p = 0; p < 2; ++p)
    a_base[p] = h + tok_of(seg, mtile + a_r + 64 * p) * Hm + a_k;
  const float* b_base = w2b + ntile + b_n;

  for (int kt = 0; kt < Hm; kt += BK) {
    // ---- stage A (h, bf16): async DMA global -> LDS ----------------------
#pragma unroll
    for (int p = 0; p < 2; ++p) {
      int row = a_r + 64 * p;
      async_copy_b128(&As[row * KP + a_k], a_base[p] + kt);
    }
    // ---- stage B (w2): two adjacent K rows, pack pairs, b32 stores -------
    // (runs while the async A copies are in flight)
#pragma unroll
    for (int p = 0; p < 2; ++p) {
      int k0 = 2 * (b_kp + 8 * p);
      const float* g2 = b_base + (long)(kt + k0) * Dm;
      float4 u = *reinterpret_cast<const float4*>(g2);
      float4 v = *reinterpret_cast<const float4*>(g2 + Dm);
      __builtin_prefetch(g2 + (long)BK * Dm, 0, 1);
      *reinterpret_cast<unsigned*>(&Bs[(b_n + 0) * KP + k0]) = pack_bf2(u.x, v.x);
      *reinterpret_cast<unsigned*>(&Bs[(b_n + 1) * KP + k0]) = pack_bf2(u.y, v.y);
      *reinterpret_cast<unsigned*>(&Bs[(b_n + 2) * KP + k0]) = pack_bf2(u.z, v.z);
      *reinterpret_cast<unsigned*>(&Bs[(b_n + 3) * KP + k0]) = pack_bf2(u.w, v.w);
    }
    async_wait0();      // A-tile DMA complete (this wave)
    __syncthreads();    // all waves' tiles visible

    v16bf af[4], bf[2];
#pragma unroll
    for (int tm = 0; tm < 4; ++tm) {
      int row = wm + 16 * tm + lr;
      af[tm] = frag16(&As[row * KP + 8 * lh], &As[row * KP + 16 + 8 * lh]);
    }
#pragma unroll
    for (int tn = 0; tn < 2; ++tn) {
      int col = wn + 16 * tn + lr;
      bf[tn] = frag16(&Bs[col * KP + 8 * lh], &Bs[col * KP + 16 + 8 * lh]);
    }
#pragma unroll
    for (int tm = 0; tm < 4; ++tm)
#pragma unroll
      for (int tn = 0; tn < 2; ++tn)
        acc[tm][tn] = __builtin_amdgcn_wmma_f32_16x16x32_bf16(
            false, af[tm], false, bf[tn], (short)0, acc[tm][tn], false, false);
    __syncthreads();
  }

#pragma unroll
  for (int tm = 0; tm < 4; ++tm)
#pragma unroll
    for (int tn = 0; tn < 2; ++tn) {
      int colg = ntile + wn + 16 * tn + lr;
#pragma unroll
      for (int r = 0; r < 8; ++r) {
        int gr = mtile + wm + 16 * tm + 8 * lh + r;
        long tok = tok_of(seg, gr);
        out[tok * Dm + colg] = acc[tm][tn][r];
      }
    }
}

// ---------------------------------------------------------------------------
extern "C" void kernel_launch(void* const* d_in, const int* in_sizes, int n_in,
                              void* d_out, int out_size, void* d_ws, size_t ws_size,
                              hipStream_t stream) {
  const float* x  = (const float*)d_in[0];
  const float* w1 = (const float*)d_in[1];
  const float* w3 = (const float*)d_in[2];
  const float* w2 = (const float*)d_in[3];
  float* out = (float*)d_out;
  unsigned short* hbuf = (unsigned short*)d_ws;  // 8*2048*2048 bf16 = 64 MiB

  dim3 blk(256);
  dim3 g1(Hm / BN, (8 * SEGL) / BM, 4);  // 16 x 32 x 4
  dim3 g2(Dm / BN, (8 * SEGL) / BM, 4);  // 8  x 32 x 4
  seg_ffn_gate_kernel<<<g1, blk, 0, stream>>>(x, w1, w3, hbuf);
  seg_ffn_down_kernel<<<g2, blk, 0, stream>>>(hbuf, w2, out);
}
